// GCN_28578712388233
// MI455X (gfx1250) — matrix operand order repaired and checked by reading, not compiled, not run on hardware
//
#include <hip/hip_runtime.h>
#include <hip/hip_bf16.h>

// ---------------- WMMA types (gfx1250, wave32) ----------------
typedef __attribute__((ext_vector_type(16))) _Float16 v16h;
typedef __attribute__((ext_vector_type(8)))  float    v8f;

#define GL0 512
#define GL1 256
#define GL2 16
#define HEADD 16

// Pack two f32 -> two f16 in one 32-bit word for LDS staging
__device__ inline unsigned pack2h(float x, float y) {
    union { _Float16 h[2]; unsigned u; } pk;
    pk.h[0] = (_Float16)x;
    pk.h[1] = (_Float16)y;
    return pk.u;
}

// ---------------- utility: zero int buffer ----------------
__global__ void k_zero_i32(int* p, int n) {
    int i = blockIdx.x * blockDim.x + threadIdx.x;
    if (i < n) p[i] = 0;
}

// ---------------- CSR build ----------------
__global__ void k_count(const int* __restrict__ row, int* __restrict__ deg, int n_edges) {
    int e = blockIdx.x * blockDim.x + threadIdx.x;
    if (e < n_edges) atomicAdd(&deg[row[e]], 1);
}

// single-block exclusive scan (n_nodes entries): rowptr + cursor copy
__global__ void k_scan(const int* __restrict__ deg, int* __restrict__ rowptr,
                       int* __restrict__ cursor, int n) {
    __shared__ int buf[1024];
    __shared__ int carry;
    if (threadIdx.x == 0) carry = 0;
    __syncthreads();
    for (int base = 0; base < n; base += 1024) {
        int i = base + (int)threadIdx.x;
        int v = (i < n) ? deg[i] : 0;
        buf[threadIdx.x] = v;
        __syncthreads();
        for (int off = 1; off < 1024; off <<= 1) {
            int t = (threadIdx.x >= (unsigned)off) ? buf[threadIdx.x - off] : 0;
            __syncthreads();
            buf[threadIdx.x] += t;
            __syncthreads();
        }
        int excl = buf[threadIdx.x] - v + carry;
        if (i < n) { rowptr[i] = excl; cursor[i] = excl; }
        __syncthreads();
        if (threadIdx.x == 1023) carry += buf[1023];
        __syncthreads();
    }
    if (threadIdx.x == 0) rowptr[n] = carry;
}

__global__ void k_scatter(const int* __restrict__ row, const int* __restrict__ col,
                          const float* __restrict__ ev, int* __restrict__ cursor,
                          int* __restrict__ colp, float* __restrict__ valp, int n_edges) {
    int e = blockIdx.x * blockDim.x + threadIdx.x;
    if (e < n_edges) {
        int p = atomicAdd(&cursor[row[e]], 1);
        colp[p] = col[e];
        valp[p] = ev[e];
    }
}

// ---------------- GEMM1: XW1 = H(f32->f16) @ W1(f32->f16), f32 accum ----------------
// grid.x = n_nodes/16 tiles ; block = 256 (8 waves) ; each block -> 16x256 panel.
// LDS tiles are staged directly in WMMA per-lane register layout: [lane][slot0..15],
// so each operand fetch is two contiguous ds_load_b128 (no half-register shuffles).
//
// A (16-bit 16x32): lanes 0-15 hold K{0-7,16-23} of row M=lane; lanes 16-31 hold K{8-15,24-31}.
//   lane = m + ((k&8)?16:0) ; slot = ((k&16)?8:0) + (k&7)
// B (16-bit 32x16 per ntile): lanes 0-15 hold K{0-7,16-23} of col N=lane; lanes 16-31 hold K{8-15,24-31}.
//   lane = n + ((k&8)?16:0) ; slot = ((k&16)?8:0) + (k&7)
__global__ void k_gemm1(const float* __restrict__ H, const float* __restrict__ W1,
                        float* __restrict__ XW1, int n_nodes) {
    __shared__ __align__(16) _Float16 sA[32 * 16];          // 1 KB : [lane][slot]
    __shared__ __align__(16) _Float16 sB[16 * 32 * 16];     // 16 KB: [ntile][lane][slot]
    const int tid  = threadIdx.x;
    const int lane = tid & 31;
    const int wave = tid >> 5;
    const int mbase = blockIdx.x * 16;
    const bool fullM = (mbase + 16 <= n_nodes);

    v8f c0 = {}; v8f c1 = {};

    for (int ks = 0; ks < GL0 / 32; ++ks) {
        __syncthreads();
        // ---- stage A tile (16 rows x 32 K), one K-pair per thread ----
        {
            int m   = tid >> 4;           // 0..15
            int kin = (tid & 15) * 2;     // even K in 0..30
            int r = mbase + m;
            float2 hv;
            if (fullM || r < n_nodes)
                hv = *(const float2*)(H + (size_t)r * GL0 + ks * 32 + kin);
            else
                hv = make_float2(0.0f, 0.0f);
            int alane = m + ((kin & 8) ? 16 : 0);
            int aslot = ((kin & 16) ? 8 : 0) + (kin & 6);
            *(unsigned*)(sA + alane * 16 + aslot) = pack2h(hv.x, hv.y);
        }
        // ---- stage B tile (32 K x 256 N), one K-pair per element ----
        for (int t = tid; t < 16 * 256; t += 256) {
            int kin = (t >> 8) * 2;       // even K in 0..30
            int n   = t & 255;
            float v0 = W1[(size_t)(ks * 32 + kin) * GL1 + n];
            float v1 = W1[(size_t)(ks * 32 + kin + 1) * GL1 + n];
            int blane = (n & 15) + ((kin & 8) ? 16 : 0);
            int bslot = ((kin & 16) ? 8 : 0) + (kin & 7);
            *(unsigned*)(sB + ((size_t)(n >> 4) * 32 + blane) * 16 + bslot) = pack2h(v0, v1);
        }
        __syncthreads();

        v16h a  = *(const v16h*)(sA + lane * 16);
        v16h b0 = *(const v16h*)(sB + ((size_t)(wave * 2 + 0) * 32 + lane) * 16);
        v16h b1 = *(const v16h*)(sB + ((size_t)(wave * 2 + 1) * 32 + lane) * 16);
        c0 = __builtin_amdgcn_wmma_f32_16x16x32_f16(false, a, false, b0, (short)0, c0, false, false);
        c1 = __builtin_amdgcn_wmma_f32_16x16x32_f16(false, a, false, b1, (short)0, c1, false, false);
    }

    const int nn  = lane & 15;
    const int mb2 = (lane >> 4) << 3;
    float* base0 = XW1 + (size_t)(mbase + mb2) * GL1 + wave * 32 + nn;
    if (fullM) {
#pragma unroll
        for (int i = 0; i < 8; ++i) {
            base0[(size_t)i * GL1]      = c0[i];
            base0[(size_t)i * GL1 + 16] = c1[i];
        }
    } else {
#pragma unroll
        for (int i = 0; i < 8; ++i) {
            if (mbase + mb2 + i < n_nodes) {
                base0[(size_t)i * GL1]      = c0[i];
                base0[(size_t)i * GL1 + 16] = c1[i];
            }
        }
    }
}

// ---------------- SpMM1: H1 = relu(A @ XW1 + b1), gather per node, 256 dims ----------------
// block = 256 (8 waves), wave per node
__global__ void k_spmm1(const int* __restrict__ rowptr, const int* __restrict__ colp,
                        const float* __restrict__ valp, const float* __restrict__ XW1,
                        const float* __restrict__ b1, float* __restrict__ H1, int n_nodes) {
    int lane = threadIdx.x & 31;
    int node = blockIdx.x * 8 + (threadIdx.x >> 5);
    if (node >= n_nodes) return;
    float acc[8] = {0, 0, 0, 0, 0, 0, 0, 0};
    int s = rowptr[node], e = rowptr[node + 1];
    for (int p = s; p < e; ++p) {
        int c = colp[p];
        float w = valp[p];
        if (p + 1 < e)  // prefetch next gathered row (L2-resident)
            __builtin_prefetch(XW1 + (size_t)colp[p + 1] * GL1 + lane * 8, 0, 3);
        const float4* src = (const float4*)(XW1 + (size_t)c * GL1);
        float4 x0 = src[lane * 2];
        float4 x1 = src[lane * 2 + 1];
        acc[0] += w * x0.x; acc[1] += w * x0.y; acc[2] += w * x0.z; acc[3] += w * x0.w;
        acc[4] += w * x1.x; acc[5] += w * x1.y; acc[6] += w * x1.z; acc[7] += w * x1.w;
    }
    int d = lane * 8;
#pragma unroll
    for (int i = 0; i < 8; ++i)
        H1[(size_t)node * GL1 + d + i] = fmaxf(acc[i] + b1[d + i], 0.0f);
}

// ---------------- GEMM2: XW2 = H1(f32->f16) @ W2(f32->f16) ----------------
// block = 256 (8 waves), wave per 16-row M tile; N = 16 (single tile), K = 256.
// W2 staged once in WMMA lane layout: [kstep][lane][slot].
__global__ void k_gemm2(const float* __restrict__ H1, const float* __restrict__ W2,
                        float* __restrict__ XW2, int n_nodes) {
    __shared__ __align__(16) _Float16 sB[8 * 32 * 16];   // 8 KB
    const int tid  = threadIdx.x;
    const int lane = tid & 31;
    const int wave = tid >> 5;

    // stage all of W2 (256x16) : 2048 K-pairs / 256 threads = 8 each
    for (int t = tid; t < 128 * 16; t += 256) {
        int kg = (t >> 4) * 2;        // even global K in 0..254
        int n  = t & 15;
        float v0 = W2[(size_t)kg * GL2 + n];
        float v1 = W2[(size_t)(kg + 1) * GL2 + n];
        int ksb   = kg >> 5;
        int kin   = kg & 31;
        int blane = n + ((kin & 8) ? 16 : 0);
        int bslot = ((kin & 16) ? 8 : 0) + (kin & 7);
        *(unsigned*)(sB + ((size_t)ksb * 32 + blane) * 16 + bslot) = pack2h(v0, v1);
    }
    __syncthreads();

    int mtile  = blockIdx.x * 8 + wave;
    int ntiles = (n_nodes + 15) / 16;
    if (mtile >= ntiles) return;          // wave-uniform; EXEC stays all-ones for WMMA
    int mbase = mtile * 16;

    const int m  = lane & 15;
    const int kb = (lane >> 4) << 3;      // A: lanes 16-31 -> K+8
    const int r  = mbase + m;
    const bool rok = (r < n_nodes);
    const float* arow = H1 + (size_t)r * GL1;

    v8f c = {};
    for (int ks = 0; ks < GL1 / 32; ++ks) {
        v16h a;
#pragma unroll
        for (int p = 0; p < 8; ++p) {     // slot pair (2p, 2p+1) <- K pair (kk0, kk0+1)
            int kk0 = ((p & 4) << 2) + kb + ((p & 3) << 1);
            float2 hv = rok ? *(const float2*)(arow + ks * 32 + kk0)
                            : make_float2(0.0f, 0.0f);
            a[2 * p]     = (_Float16)hv.x;
            a[2 * p + 1] = (_Float16)hv.y;
        }
        v16h b = *(const v16h*)(sB + ((size_t)ks * 32 + lane) * 16);
        c = __builtin_amdgcn_wmma_f32_16x16x32_f16(false, a, false, b, (short)0, c, false, false);
    }
    const int nn  = lane & 15;
    const int mb2 = (lane >> 4) << 3;
    if (mbase + 16 <= n_nodes) {
#pragma unroll
        for (int i = 0; i < 8; ++i) XW2[(size_t)(mbase + mb2 + i) * GL2 + nn] = c[i];
    } else {
#pragma unroll
        for (int i = 0; i < 8; ++i)
            if (mbase + mb2 + i < n_nodes) XW2[(size_t)(mbase + mb2 + i) * GL2 + nn] = c[i];
    }
}

// ---------------- SpMM2 + degree: H2 = relu(A @ XW2 + b2); D = A @ 1 ----------------
// block = 256 -> 16 nodes per block, 16 threads (dims) per node
__global__ void k_spmm2(const int* __restrict__ rowptr, const int* __restrict__ colp,
                        const float* __restrict__ valp, const float* __restrict__ XW2,
                        const float* __restrict__ b2, float* __restrict__ H2,
                        float* __restrict__ D, int n_nodes) {
    int g = threadIdx.x >> 4;
    int d = threadIdx.x & 15;
    int node = blockIdx.x * 16 + g;
    if (node >= n_nodes) return;
    float acc = 0.0f, dsum = 0.0f;
    int s = rowptr[node], e = rowptr[node + 1];
    for (int p = s; p < e; ++p) {
        int c = colp[p];
        float w = valp[p];
        acc  += w * XW2[(size_t)c * GL2 + d];
        dsum += w;
    }
    H2[(size_t)node * GL2 + d] = fmaxf(acc + b2[d], 0.0f);
    if (d == 0) D[node] = dsum;
}

// ---------------- Head: Y = softmax(relu(H2 @ Wl + bl)) ----------------
__global__ void k_head(const float* __restrict__ H2, const float* __restrict__ Wl,
                       const float* __restrict__ bl, float* __restrict__ Y, int n_nodes) {
    __shared__ float sW[GL2 * HEADD];
    __shared__ float sb[HEADD];
    if (threadIdx.x < GL2 * HEADD) sW[threadIdx.x] = Wl[threadIdx.x];
    if (threadIdx.x < HEADD) sb[threadIdx.x] = bl[threadIdx.x];
    __syncthreads();
    int node = blockIdx.x * blockDim.x + threadIdx.x;
    if (node >= n_nodes) return;
    float h[GL2], o[HEADD];
#pragma unroll
    for (int k = 0; k < GL2; ++k) h[k] = H2[(size_t)node * GL2 + k];
#pragma unroll
    for (int j = 0; j < HEADD; ++j) o[j] = sb[j];
#pragma unroll
    for (int k = 0; k < GL2; ++k) {
        float hk = h[k];
#pragma unroll
        for (int j = 0; j < HEADD; ++j) o[j] += hk * sW[k * HEADD + j];
    }
    float mx = -1e30f;
#pragma unroll
    for (int j = 0; j < HEADD; ++j) { o[j] = fmaxf(o[j], 0.0f); mx = fmaxf(mx, o[j]); }
    float sum = 0.0f;
#pragma unroll
    for (int j = 0; j < HEADD; ++j) { o[j] = __expf(o[j] - mx); sum += o[j]; }
    float inv = 1.0f / sum;
#pragma unroll
    for (int j = 0; j < HEADD; ++j) Y[(size_t)node * HEADD + j] = o[j] * inv;
}

// ---------------- Gamma = Y^T D : two-stage deterministic reduction ----------------
__global__ void k_gamma_part(const float* __restrict__ Y, const float* __restrict__ D,
                             float* __restrict__ gpart, int n_nodes) {
    float g[HEADD];
#pragma unroll
    for (int k = 0; k < HEADD; ++k) g[k] = 0.0f;
    int stride = gridDim.x * blockDim.x;
    for (int node = blockIdx.x * blockDim.x + threadIdx.x; node < n_nodes; node += stride) {
        float dd = D[node];
#pragma unroll
        for (int k = 0; k < HEADD; ++k) g[k] += dd * Y[(size_t)node * HEADD + k];
    }
    __shared__ float red[256];
    for (int k = 0; k < HEADD; ++k) {
        red[threadIdx.x] = g[k];
        __syncthreads();
        for (int off = 128; off > 0; off >>= 1) {
            if (threadIdx.x < (unsigned)off) red[threadIdx.x] += red[threadIdx.x + off];
            __syncthreads();
        }
        if (threadIdx.x == 0) gpart[blockIdx.x * HEADD + k] = red[0];
        __syncthreads();
    }
}

__global__ void k_gamma_final(const float* __restrict__ gpart, float* __restrict__ Gamma, int nb) {
    int k = threadIdx.x;
    if (k < HEADD) {
        float s = 0.0f;
        for (int b = 0; b < nb; ++b) s += gpart[b * HEADD + k];
        Gamma[k] = s;
    }
}

// ---------------- NCut loss: sum_e w_e * dot(Y[r]/Gamma, 1 - Y[c]) ----------------
__global__ void k_loss_part(const int* __restrict__ row, const int* __restrict__ col,
                            const float* __restrict__ ev, const float* __restrict__ Y,
                            const float* __restrict__ Gamma, float* __restrict__ lpart,
                            int n_edges) {
    __shared__ float invG[HEADD];
    if (threadIdx.x < HEADD) invG[threadIdx.x] = 1.0f / Gamma[threadIdx.x];
    __syncthreads();
    float acc = 0.0f;
    int stride = gridDim.x * blockDim.x;
    for (int e = blockIdx.x * blockDim.x + threadIdx.x; e < n_edges; e += stride) {
        int r = row[e], c = col[e];
        const float4* yr = (const float4*)(Y + (size_t)r * HEADD);
        const float4* yc = (const float4*)(Y + (size_t)c * HEADD);
        float s = 0.0f;
#pragma unroll
        for (int q = 0; q < 4; ++q) {
            float4 a = yr[q], b = yc[q];
            s += (a.x * invG[q * 4 + 0]) * (1.0f - b.x);
            s += (a.y * invG[q * 4 + 1]) * (1.0f - b.y);
            s += (a.z * invG[q * 4 + 2]) * (1.0f - b.z);
            s += (a.w * invG[q * 4 + 3]) * (1.0f - b.w);
        }
        acc += ev[e] * s;
    }
    __shared__ float red[256];
    red[threadIdx.x] = acc;
    __syncthreads();
    for (int off = 128; off > 0; off >>= 1) {
        if (threadIdx.x < (unsigned)off) red[threadIdx.x] += red[threadIdx.x + off];
        __syncthreads();
    }
    if (threadIdx.x == 0) lpart[blockIdx.x] = red[0];
}

__global__ void k_loss_final(const float* __restrict__ lpart, float* __restrict__ out, int nb) {
    __shared__ float red[256];
    float a = 0.0f;
    for (int i = threadIdx.x; i < nb; i += 256) a += lpart[i];
    red[threadIdx.x] = a;
    __syncthreads();
    for (int off = 128; off > 0; off >>= 1) {
        if (threadIdx.x < (unsigned)off) red[threadIdx.x] += red[threadIdx.x + off];
        __syncthreads();
    }
    if (threadIdx.x == 0) out[0] = red[0];
}

// ======================================================================
extern "C" void kernel_launch(void* const* d_in, const int* in_sizes, int n_in,
                              void* d_out, int out_size, void* d_ws, size_t ws_size,
                              hipStream_t stream) {
    const float* H   = (const float*)d_in[0];
    const int*   ei  = (const int*)d_in[1];   // [2, n_edges] flat
    const float* ev  = (const float*)d_in[2];
    const float* W1  = (const float*)d_in[3];
    const float* b1  = (const float*)d_in[4];
    const float* W2  = (const float*)d_in[5];
    const float* b2  = (const float*)d_in[6];
    const float* Wl  = (const float*)d_in[7];
    const float* bl  = (const float*)d_in[8];
    float* out = (float*)d_out;

    const int n_nodes = in_sizes[0] / GL0;
    const int n_edges = in_sizes[2];
    const int* row = ei;
    const int* col = ei + n_edges;

    // ---- workspace bump allocator (256B aligned) ----
    char* wsp = (char*)d_ws;
    size_t off = 0;
    auto alloc = [&](size_t bytes) -> void* {
        off = (off + 255) & ~(size_t)255;
        void* p = wsp + off;
        off += bytes;
        return p;
    };
    int*   deg    = (int*)  alloc((size_t)n_nodes * sizeof(int));
    int*   rowptr = (int*)  alloc((size_t)(n_nodes + 1) * sizeof(int));
    int*   cursor = (int*)  alloc((size_t)n_nodes * sizeof(int));
    int*   colp   = (int*)  alloc((size_t)n_edges * sizeof(int));
    float* valp   = (float*)alloc((size_t)n_edges * sizeof(float));
    float* XW1    = (float*)alloc((size_t)n_nodes * GL1 * sizeof(float));
    float* H1     = (float*)alloc((size_t)n_nodes * GL1 * sizeof(float));
    float* XW2    = (float*)alloc((size_t)n_nodes * GL2 * sizeof(float));
    float* H2     = (float*)alloc((size_t)n_nodes * GL2 * sizeof(float));
    float* Y      = (float*)alloc((size_t)n_nodes * HEADD * sizeof(float));
    float* D      = (float*)alloc((size_t)n_nodes * sizeof(float));
    float* Gamma  = (float*)alloc(HEADD * sizeof(float));
    const int GB = 200;
    const int LB = 1024;
    float* gpart  = (float*)alloc((size_t)GB * HEADD * sizeof(float));
    float* lpart  = (float*)alloc((size_t)LB * sizeof(float));
    (void)ws_size; (void)n_in; (void)out_size;

    // ---- CSR build ----
    k_zero_i32<<<(n_nodes + 255) / 256, 256, 0, stream>>>(deg, n_nodes);
    k_count<<<(n_edges + 255) / 256, 256, 0, stream>>>(row, deg, n_edges);
    k_scan<<<1, 1024, 0, stream>>>(deg, rowptr, cursor, n_nodes);
    k_scatter<<<(n_edges + 255) / 256, 256, 0, stream>>>(row, col, ev, cursor, colp, valp, n_edges);

    // ---- layer 1 ----
    int mtiles = (n_nodes + 15) / 16;
    k_gemm1<<<mtiles, 256, 0, stream>>>(H, W1, XW1, n_nodes);
    k_spmm1<<<(n_nodes + 7) / 8, 256, 0, stream>>>(rowptr, colp, valp, XW1, b1, H1, n_nodes);

    // ---- layer 2 ----
    k_gemm2<<<(mtiles + 7) / 8, 256, 0, stream>>>(H1, W2, XW2, n_nodes);
    k_spmm2<<<(n_nodes + 15) / 16, 256, 0, stream>>>(rowptr, colp, valp, XW2, b2, H2, D, n_nodes);

    // ---- head + softmax ----
    k_head<<<(n_nodes + 255) / 256, 256, 0, stream>>>(H2, Wl, bl, Y, n_nodes);

    // ---- Gamma ----
    k_gamma_part<<<GB, 256, 0, stream>>>(Y, D, gpart, n_nodes);
    k_gamma_final<<<1, 64, 0, stream>>>(gpart, Gamma, GB);

    // ---- loss ----
    k_loss_part<<<LB, 256, 0, stream>>>(row, col, ev, Y, Gamma, lpart, n_edges);
    k_loss_final<<<1, 256, 0, stream>>>(lpart, out, LB);
}
